// MultiHeadSelfAttention_73967926772427
// MI455X (gfx1250) — matrix operand order repaired and checked by reading, not compiled, not run on hardware
//
#include <hip/hip_runtime.h>

#define EMBED 1024
#define NHEAD 16
#define HDIM  64
#define SEQ   2048
#define BATCH 2
#define ROWS  (BATCH * SEQ)   // 4096
#define QKVC  (3 * EMBED)     // 3072
#define PLD   40              // padded LDS row stride (f16 elems) for P tile
#define NK    (EMBED / 32)    // 32 k-steps in the projection GEMMs

typedef __attribute__((ext_vector_type(16))) _Float16 v16h;
typedef __attribute__((ext_vector_type(8)))  _Float16 v8h;
typedef __attribute__((ext_vector_type(8)))  float    v8f;
typedef __attribute__((ext_vector_type(4)))  unsigned int v4u;
typedef __attribute__((ext_vector_type(8)))  int v8i;
typedef __attribute__((ext_vector_type(4)))  int v4i;

#if defined(__has_builtin)
#  if __has_builtin(__builtin_amdgcn_tensor_load_to_lds)
#    define HAVE_TDM 1
#    if defined(__has_include)
#      if __has_include(<hip/amd_detail/amd_gfx1250_TDM.h>)
#        define TDM_6ARG 1
#      endif
#    endif
#  endif
#endif

static __device__ __forceinline__ v8f wmma_f16(v16h a, v16h b, v8f c) {
  return __builtin_amdgcn_wmma_f32_16x16x32_f16(false, a, false, b, (short)0, c,
                                                false, false);
}

#ifdef HAVE_TDM
// TDM 2D tile load: f16 tensor, row-major, tile (tile_w x tile_h) at gsrc with row
// stride stride_e (elements), packed contiguously into LDS at lds_off.
// D# layout per CDNA5 ISA. No wait here (async).
static __device__ __forceinline__ void tdm_load_2d(const void* gsrc, unsigned lds_off,
                                                   unsigned tens_w, unsigned tens_h,
                                                   unsigned tile_w, unsigned tile_h,
                                                   unsigned stride_e) {
  const unsigned long long ga = (unsigned long long)(size_t)gsrc;
  const unsigned long long s0 = (unsigned long long)stride_e;
  v4u g0;
  g0[0] = 1u;                                           // count=1 (valid), user mode
  g0[1] = lds_off;                                      // lds_addr (bytes)
  g0[2] = (unsigned)(ga & 0xffffffffull);               // global_addr[31:0]
  g0[3] = (unsigned)((ga >> 32) & 0x01ffffffull) | (2u << 30);  // [56:32] | type=2
  v8i g1;
  g1[0] = (int)(1u << 16);                              // data_size=1 (2 bytes)
  g1[1] = (int)(tens_w << 16);                          // tensor_dim0[15:0]
  g1[2] = (int)((tens_w >> 16) & 0xffffu) | (int)(tens_h << 16);
  g1[3] = (int)((tens_h >> 16) & 0xffffu) | (int)(tile_w << 16);
  g1[4] = (int)(tile_h & 0xffffu);                      // tile_dim1; tile_dim2=0
  g1[5] = (int)(unsigned)(s0 & 0xffffffffull);          // dim0_stride[31:0]
  g1[6] = (int)(unsigned)((s0 >> 32) & 0xffffull);      // dim0_stride[47:32]
  g1[7] = 0;
  const v4i z4 = {0, 0, 0, 0};
#ifdef TDM_6ARG
  const v8i z8 = {0, 0, 0, 0, 0, 0, 0, 0};
  __builtin_amdgcn_tensor_load_to_lds(g0, g1, z4, z4, z8, 0);
#else
  __builtin_amdgcn_tensor_load_to_lds(g0, g1, z4, z4, 0);
#endif
}
#endif

// A fragment (16x32 f16): rows row0..+15, k cols k0..+31 of row-major [*, ld].
static __device__ __forceinline__ v16h load_a(const _Float16* p, int ld, int row0, int k0) {
  const int lane = threadIdx.x & 31;
  const _Float16* r = p + (size_t)(row0 + (lane & 15)) * ld + k0 + ((lane >> 4) << 3);
  const v8h lo = *(const v8h*)r;
  const v8h hi = *(const v8h*)(r + 16);
  v16h a;
#pragma unroll
  for (int e = 0; e < 8; ++e) { a[e] = lo[e]; a[8 + e] = hi[e]; }
  return a;
}

// B fragment (32x16) as transpose of a row-major tile: B[k][n] = p[row0+n][k0+k].
static __device__ __forceinline__ v16h load_bT(const _Float16* p, int ld, int row0, int k0) {
  const int lane = threadIdx.x & 31;
  const _Float16* r = p + (size_t)(row0 + (lane & 15)) * ld + k0 + ((lane >> 4) << 3);
  const v8h lo = *(const v8h*)r;
  const v8h hi = *(const v8h*)(r + 16);
  v16h b;
#pragma unroll
  for (int e = 0; e < 8; ++e) { b[e] = lo[e]; b[8 + e] = hi[e]; }
  return b;
}

// ---- GEMM tile staging (A 128x32, Bt 64x32) ----
#ifdef HAVE_TDM
static __device__ __forceinline__ void gemm_stage_issue(const _Float16* A,
                                                        const _Float16* Bt,
                                                        _Float16* dA, _Float16* dB,
                                                        int rowB, int col0, int k0,
                                                        int tensH_A, int tensH_B) {
  tdm_load_2d(A + (size_t)rowB * EMBED + k0, (unsigned)(size_t)(void*)dA,
              EMBED, (unsigned)tensH_A, 32, 128, EMBED);
  tdm_load_2d(Bt + (size_t)col0 * EMBED + k0, (unsigned)(size_t)(void*)dB,
              EMBED, (unsigned)tensH_B, 32, 64, EMBED);
}
#else
static __device__ __forceinline__ void gemm_stage_copy(const _Float16* A,
                                                       const _Float16* Bt,
                                                       _Float16* dA, _Float16* dB,
                                                       int rowB, int col0, int k0) {
  for (int idx = threadIdx.x * 8; idx < 128 * 32; idx += 256 * 8)
    *(v8h*)(dA + idx) =
        *(const v8h*)(A + (size_t)(rowB + (idx >> 5)) * EMBED + k0 + (idx & 31));
  for (int idx = threadIdx.x * 8; idx < 64 * 32; idx += 256 * 8)
    *(v8h*)(dB + idx) =
        *(const v8h*)(Bt + (size_t)(col0 + (idx >> 5)) * EMBED + k0 + (idx & 31));
}
#endif

__global__ void k_cvt(const float* __restrict__ in, _Float16* __restrict__ out, int n) {
  int i = blockIdx.x * blockDim.x + threadIdx.x;
  if (i < n) out[i] = (_Float16)in[i];
}

// Transpose + convert: in f32 [K,N] -> out f16 [N,K]
__global__ void k_cvt_t(const float* __restrict__ in, _Float16* __restrict__ out,
                        int K, int N) {
  int i = blockIdx.x * blockDim.x + threadIdx.x;
  if (i < K * N) {
    int k = i / N, n = i - k * N;
    out[(size_t)n * K + k] = (_Float16)in[i];
  }
}

// Double-buffered TDM mainloop shared by both GEMMs (K = EMBED for both).
// Block = 8 waves = 128 rows x 64 cols; `wid` is forced scalar (readfirstlane) so
// the issuing-wave branch is an SCC branch and descriptors build in SALU.
#define GEMM_MAINLOOP(A_, Bt_, tensH_A_, tensH_B_)                                   \
  __shared__ __attribute__((aligned(16))) _Float16 sA[2][128 * 32];                  \
  __shared__ __attribute__((aligned(16))) _Float16 sB[2][64 * 32];                   \
  const int lane = threadIdx.x & 31;                                                 \
  const int wid  = __builtin_amdgcn_readfirstlane(threadIdx.x >> 5);                 \
  const int rowB = blockIdx.y * 128;                                                 \
  const int col0 = blockIdx.x * 64;                                                  \
  const int row0 = rowB + wid * 16;                                                  \
  v8f acc0 = {}, acc1 = {}, acc2 = {}, acc3 = {};                                    \
  GEMM_PROLOGUE(A_, Bt_, tensH_A_, tensH_B_);                                        \
  __syncthreads();                                                                   \
  for (int ks = 0; ks < NK; ++ks) {                                                  \
    const int cur = ks & 1;                                                          \
    GEMM_STAGE_NEXT(A_, Bt_, tensH_A_, tensH_B_);                                    \
    v16h a = load_a(sA[cur], 32, wid * 16, 0);                                       \
    acc0 = wmma_f16(a, load_bT(sB[cur], 32, 0,  0), acc0);                           \
    acc1 = wmma_f16(a, load_bT(sB[cur], 32, 16, 0), acc1);                           \
    acc2 = wmma_f16(a, load_bT(sB[cur], 32, 32, 0), acc2);                           \
    acc3 = wmma_f16(a, load_bT(sB[cur], 32, 48, 0), acc3);                           \
    GEMM_STAGE_WAIT();                                                               \
    __syncthreads();                                                                 \
  }                                                                                  \
  v8f acc[4] = {acc0, acc1, acc2, acc3};

#ifdef HAVE_TDM
#define GEMM_PROLOGUE(A_, Bt_, hA_, hB_)                                             \
  if (wid == 0)                                                                      \
    gemm_stage_issue(A_, Bt_, sA[0], sB[0], rowB, col0, 0, hA_, hB_);                \
  __builtin_amdgcn_s_wait_tensorcnt(0);
#define GEMM_STAGE_NEXT(A_, Bt_, hA_, hB_)                                           \
  if (wid == 0 && ks + 1 < NK)                                                       \
    gemm_stage_issue(A_, Bt_, sA[cur ^ 1], sB[cur ^ 1], rowB, col0, (ks + 1) * 32,   \
                     hA_, hB_);
#define GEMM_STAGE_WAIT() __builtin_amdgcn_s_wait_tensorcnt(0);
#else
#define GEMM_PROLOGUE(A_, Bt_, hA_, hB_) gemm_stage_copy(A_, Bt_, sA[0], sB[0], rowB, col0, 0)
#define GEMM_STAGE_NEXT(A_, Bt_, hA_, hB_)                                           \
  if (ks + 1 < NK)                                                                   \
    gemm_stage_copy(A_, Bt_, sA[cur ^ 1], sB[cur ^ 1], rowB, col0, (ks + 1) * 32);
#define GEMM_STAGE_WAIT()
#endif

// QKV GEMM: Q,K columns -> qkv row-major [ROWS,3072]; V -> vt [B][H][D][T].
__global__ __launch_bounds__(256) void k_gemm_qkv(const _Float16* __restrict__ A,
                                                  const _Float16* __restrict__ Bt,
                                                  const float* __restrict__ bias,
                                                  _Float16* __restrict__ qkv,
                                                  _Float16* __restrict__ vt) {
  GEMM_MAINLOOP(A, Bt, ROWS, QKVC)
  const int n = lane & 15, hi = lane >> 4;
#pragma unroll
  for (int t = 0; t < 4; ++t) {
    const int c = col0 + 16 * t + n;
    const float bv = bias[c];
    if (c < 2 * EMBED) {
#pragma unroll
      for (int j = 0; j < 8; ++j)
        qkv[(size_t)(row0 + j + 8 * hi) * QKVC + c] = (_Float16)(acc[t][j] + bv);
    } else {                       // V: transposed store, contiguous along T (b128)
      const int b  = row0 >> 11;
      const int t0 = (row0 & (SEQ - 1)) + 8 * hi;
      const int cd = c - 2 * EMBED;
      v8h vv;
#pragma unroll
      for (int j = 0; j < 8; ++j) vv[j] = (_Float16)(acc[t][j] + bv);
      *(v8h*)(vt + ((size_t)(b * EMBED + cd)) * SEQ + t0) = vv;
    }
  }
}

// Output GEMM: out f32 = attn16 * W_out + b.
__global__ __launch_bounds__(256) void k_gemm_out(const _Float16* __restrict__ A,
                                                  const _Float16* __restrict__ Bt,
                                                  const float* __restrict__ bias,
                                                  float* __restrict__ out) {
  GEMM_MAINLOOP(A, Bt, ROWS, EMBED)
  const int n = lane & 15, hi = lane >> 4;
#pragma unroll
  for (int t = 0; t < 4; ++t) {
    const int c = col0 + 16 * t + n;
    const float bv = bias[c];
#pragma unroll
    for (int j = 0; j < 8; ++j)
      out[(size_t)(row0 + j + 8 * hi) * EMBED + c] = acc[t][j] + bv;
  }
}

// ---- attention K/V staging: K tile 32x64 (keys x d), V tile 64x32 (d x keys) ----
static __device__ __forceinline__ void attn_stage(const _Float16* Kb, const _Float16* Vt,
                                                  _Float16* dK, _Float16* dV, int key0) {
#ifdef HAVE_TDM
  tdm_load_2d(Kb + (size_t)key0 * QKVC, (unsigned)(size_t)(void*)dK,
              HDIM, SEQ, HDIM, 32, QKVC);
  tdm_load_2d(Vt + key0, (unsigned)(size_t)(void*)dV,
              SEQ, HDIM, 32, HDIM, SEQ);
#else
  for (int idx = (threadIdx.x & 31) * 8; idx < 32 * HDIM; idx += 32 * 8)
    *(v8h*)(dK + idx) =
        *(const v8h*)(Kb + (size_t)(key0 + (idx >> 6)) * QKVC + (idx & 63));
  for (int idx = (threadIdx.x & 31) * 8; idx < HDIM * 32; idx += 32 * 8)
    *(v8h*)(dV + idx) =
        *(const v8h*)(Vt + (size_t)(idx >> 5) * SEQ + key0 + (idx & 31));
#endif
}

// Flash attention: block = one (b,h) x 8 consecutive 16-query tiles (8 waves).
// K/V tiles TDM-staged into LDS once per block (8x reuse), double-buffered.
__global__ __launch_bounds__(256) void k_attn(const _Float16* __restrict__ qkv,
                                              const _Float16* __restrict__ vt,
                                              _Float16* __restrict__ attn) {
  __shared__ __attribute__((aligned(16))) _Float16 sK[2][32 * HDIM];
  __shared__ __attribute__((aligned(16))) _Float16 sV[2][HDIM * 32];
  __shared__ __attribute__((aligned(16))) _Float16 pbuf[8][16 * PLD];
  const int lane = threadIdx.x & 31;
  const int wid  = __builtin_amdgcn_readfirstlane(threadIdx.x >> 5);  // scalar
  const int qt0  = (blockIdx.x & 15) * 8;        // 16 q-tile groups of 8
  const int h    = (blockIdx.x >> 4) & 15;
  const int b    = blockIdx.x >> 8;
  const int qt   = qt0 + wid;

  const int brow = b * SEQ;
  const _Float16* Qb = qkv + (size_t)brow * QKVC + h * HDIM;
  const _Float16* Kb = Qb + EMBED;
  const _Float16* Vt = vt + ((size_t)(b * NHEAD + h) * HDIM) * SEQ;

  v16h qa0 = load_a(Qb, QKVC, qt * 16, 0);
  v16h qa1 = load_a(Qb, QKVC, qt * 16, 32);
#pragma unroll
  for (int e = 0; e < 16; ++e) {   // fold 1/sqrt(64) into Q (exact in f16)
    qa0[e] *= (_Float16)0.125f;
    qa1[e] *= (_Float16)0.125f;
  }

  float m[8], l[8];
#pragma unroll
  for (int j = 0; j < 8; ++j) { m[j] = -1e30f; l[j] = 0.0f; }
  v8f o0 = {}, o1 = {}, o2 = {}, o3 = {};

  const int n = lane & 15, hi = lane >> 4;
  const int qrow0 = qt * 16 + 8 * hi;
  _Float16* Pw = &pbuf[wid][0];

  const int ksmax_w = (qt * 16 + 15) >> 5;             // scalar: wave causal limit
  const int ksmax_b = ((qt0 + 7) * 16 + 15) >> 5;      // block's limit

  if (wid == 0) attn_stage(Kb, Vt, sK[0], sV[0], 0);
#ifdef HAVE_TDM
  __builtin_amdgcn_s_wait_tensorcnt(0);
#endif
  __syncthreads();

  for (int ks = 0; ks <= ksmax_b; ++ks) {
    const int cur = ks & 1;
    if (wid == 0 && ks < ksmax_b)
      attn_stage(Kb, Vt, sK[cur ^ 1], sV[cur ^ 1], (ks + 1) * 32);

    if (ks <= ksmax_w) {                                // scalar branch per wave
      const int key0 = ks * 32;
      v8f s0 = {}, s1 = {};
      s0 = wmma_f16(qa0, load_bT(sK[cur], HDIM, 0, 0),   s0);
      s0 = wmma_f16(qa1, load_bT(sK[cur], HDIM, 0, 32),  s0);
      s1 = wmma_f16(qa0, load_bT(sK[cur], HDIM, 16, 0),  s1);
      s1 = wmma_f16(qa1, load_bT(sK[cur], HDIM, 16, 32), s1);

#pragma unroll
      for (int j = 0; j < 8; ++j) {                     // causal mask
        const int q = qrow0 + j;
        if (key0 + n > q)      s0[j] = -1e30f;
        if (key0 + 16 + n > q) s1[j] = -1e30f;
      }

      float tmax[8], p0[8], p1[8], rs[8], sc[8];
#pragma unroll
      for (int j = 0; j < 8; ++j) tmax[j] = fmaxf(s0[j], s1[j]);
      for (int mk = 1; mk <= 8; mk <<= 1)               // 16-lane row groups (wave32)
#pragma unroll
        for (int j = 0; j < 8; ++j) tmax[j] = fmaxf(tmax[j], __shfl_xor(tmax[j], mk, 32));
#pragma unroll
      for (int j = 0; j < 8; ++j) {
        const float mn = fmaxf(m[j], tmax[j]);
        sc[j] = __expf(m[j] - mn);
        p0[j] = __expf(s0[j] - mn);
        p1[j] = __expf(s1[j] - mn);
        rs[j] = p0[j] + p1[j];
        m[j]  = mn;
      }
      for (int mk = 1; mk <= 8; mk <<= 1)
#pragma unroll
        for (int j = 0; j < 8; ++j) rs[j] += __shfl_xor(rs[j], mk, 32);
#pragma unroll
      for (int j = 0; j < 8; ++j) {
        l[j] = l[j] * sc[j] + rs[j];
        o0[j] *= sc[j]; o1[j] *= sc[j]; o2[j] *= sc[j]; o3[j] *= sc[j];
      }

      // P: C-frag -> LDS -> A-frag (wave-private; DS in-order per wave)
#pragma unroll
      for (int j = 0; j < 8; ++j) {
        const int r = j + 8 * hi;
        Pw[r * PLD + n]      = (_Float16)p0[j];
        Pw[r * PLD + 16 + n] = (_Float16)p1[j];
      }
      v16h pa = load_a(Pw, PLD, 0, 0);

      o0 = wmma_f16(pa, load_bT(sV[cur], 32, 0,  0), o0);
      o1 = wmma_f16(pa, load_bT(sV[cur], 32, 16, 0), o1);
      o2 = wmma_f16(pa, load_bT(sV[cur], 32, 32, 0), o2);
      o3 = wmma_f16(pa, load_bT(sV[cur], 32, 48, 0), o3);
    }

#ifdef HAVE_TDM
    __builtin_amdgcn_s_wait_tensorcnt(0);
#endif
    __syncthreads();
  }

#pragma unroll
  for (int j = 0; j < 8; ++j) {
    const float inv = 1.0f / l[j];
    const size_t base = (size_t)(brow + qrow0 + j) * EMBED + h * HDIM + n;
    attn[base + 0]  = (_Float16)(o0[j] * inv);
    attn[base + 16] = (_Float16)(o1[j] * inv);
    attn[base + 32] = (_Float16)(o2[j] * inv);
    attn[base + 48] = (_Float16)(o3[j] * inv);
  }
}

extern "C" void kernel_launch(void* const* d_in, const int* in_sizes, int n_in,
                              void* d_out, int out_size, void* d_ws, size_t ws_size,
                              hipStream_t stream) {
  (void)in_sizes; (void)n_in; (void)out_size; (void)ws_size;
  const float* x     = (const float*)d_in[0];
  const float* W_qkv = (const float*)d_in[1];
  const float* b_qkv = (const float*)d_in[2];
  const float* W_out = (const float*)d_in[3];
  const float* b_out = (const float*)d_in[4];
  float* out = (float*)d_out;

  char* ws = (char*)d_ws;
  size_t off = 0;
  auto alloc16 = [&](size_t elems) {
    _Float16* p = (_Float16*)(ws + off);
    off += ((elems * sizeof(_Float16) + 255) / 256) * 256;
    return p;
  };
  _Float16* x16    = alloc16((size_t)ROWS * EMBED);
  _Float16* wqT    = alloc16((size_t)QKVC * EMBED);
  _Float16* woT    = alloc16((size_t)EMBED * EMBED);
  _Float16* qkv16  = alloc16((size_t)ROWS * QKVC);
  _Float16* vt     = alloc16((size_t)BATCH * EMBED * SEQ);
  _Float16* attn16 = alloc16((size_t)ROWS * EMBED);

  k_cvt  <<<dim3((ROWS * EMBED + 255) / 256), 256, 0, stream>>>(x, x16, ROWS * EMBED);
  k_cvt_t<<<dim3((EMBED * QKVC + 255) / 256), 256, 0, stream>>>(W_qkv, wqT, EMBED, QKVC);
  k_cvt_t<<<dim3((EMBED * EMBED + 255) / 256), 256, 0, stream>>>(W_out, woT, EMBED, EMBED);

  k_gemm_qkv<<<dim3(QKVC / 64, ROWS / 128), 256, 0, stream>>>(
      x16, wqT, b_qkv, qkv16, vt);

  // 512 blocks: (b, h, 8-q-tile group); 8 waves of 256 threads each
  k_attn<<<dim3(BATCH * NHEAD * 16), 256, 0, stream>>>(qkv16, vt, attn16);

  k_gemm_out<<<dim3(EMBED / 64, ROWS / 128), 256, 0, stream>>>(
      attn16, woT, b_out, out);
}
